// SPRGraphNet_88648124990950
// MI455X (gfx1250) — matrix operand order repaired and checked by reading, not compiled, not run on hardware
//
#include <hip/hip_runtime.h>

typedef __attribute__((ext_vector_type(16))) _Float16 v16h;
typedef __attribute__((ext_vector_type(8)))  float    v8f;

#define NN    100000   // nodes
#define NE    1600000  // edges
#define NG    512      // graphs
#define NCLS  10

static __device__ __forceinline__ v8f wmma16(v16h a, v16h b, v8f c) {
  // D = A(16x32 f16) * B(32x16 f16) + C(16x16 f32)
  return __builtin_amdgcn_wmma_f32_16x16x32_f16(false, a, false, b, (short)0, c,
                                                false, false);
}

// ---------------- utility ----------------
__global__ void k_zero(float* __restrict__ p, int n) {
  int i = blockIdx.x * blockDim.x + threadIdx.x;
  if (i < n) p[i] = 0.f;
}

// Pack a row-major 64x64 fp32 weight into WMMA B-fragment order (f16):
// dst[((kstep*4+ntile)*32 + lane)*16 + j] = W[kstep*32 + (j&8)*2 + (lane>=16?8:0)+(j&7)][ntile*16+(lane&15)]
__global__ void k_pack64(const float* __restrict__ w, _Float16* __restrict__ dst) {
  int idx = blockIdx.x * blockDim.x + threadIdx.x;   // 4096 total
  if (idx >= 4096) return;
  int j    = idx & 15;
  int lane = (idx >> 4) & 31;
  int nt   = (idx >> 9) & 3;
  int ks   = idx >> 11;
  int K = ks * 32 + (j & 8) * 2 + ((lane & 16) ? 8 : 0) + (j & 7);
  int c = nt * 16 + (lane & 15);
  dst[idx] = (_Float16)w[K * 64 + c];
}

// Pack 64x10 classifier weight, zero-padded to 64x16, one n-tile.
__global__ void k_packC(const float* __restrict__ w, _Float16* __restrict__ dst) {
  int idx = blockIdx.x * blockDim.x + threadIdx.x;   // 1024 total
  if (idx >= 1024) return;
  int j    = idx & 15;
  int lane = (idx >> 4) & 31;
  int ks   = idx >> 9;
  int K = ks * 32 + (j & 8) * 2 + ((lane & 16) ? 8 : 0) + (j & 7);
  int c = lane & 15;
  dst[idx] = (c < NCLS) ? (_Float16)w[K * NCLS + c] : (_Float16)0.f;
}

// h0[node][0:32] = shape_emb[x[node,0]], h0[node][32:64] = color_emb[x[node,1]]
__global__ void k_embed(const int* __restrict__ x, const float* __restrict__ se,
                        const float* __restrict__ ce, float* __restrict__ h) {
  int i = blockIdx.x * blockDim.x + threadIdx.x;
  if (i >= NN * 64) return;
  int node = i >> 6, f = i & 63;
  float v = (f < 32) ? se[x[node * 2 + 0] * 32 + f]
                     : ce[x[node * 2 + 1] * 32 + (f - 32)];
  h[i] = v;
}

// edge scatter: agg[tgt] += h[src]; cnt[tgt] += 1. One lane = (edge, 4 features):
// b128 gather of h[src], 4 fp32 atomics that resolve in the 192MB L2 (agg = 25.6MB).
__global__ void k_scatter(const float* __restrict__ h, const int* __restrict__ src,
                          const int* __restrict__ tgt, float* __restrict__ agg,
                          float* __restrict__ cnt) {
  int i = blockIdx.x * blockDim.x + threadIdx.x;  // NE*16 = 25,600,000
  if (i >= NE * 16) return;
  int e = i >> 4, q = i & 15;
  int s = src[e], t = tgt[e];
  float4 v = ((const float4*)h)[s * 16 + q];
  float* a = &agg[t * 64 + q * 4];
  atomicAdd(a + 0, v.x);
  atomicAdd(a + 1, v.y);
  atomicAdd(a + 2, v.z);
  atomicAdd(a + 3, v.w);
  if (q == 0) atomicAdd(&cnt[t], 1.f);
}

// Fused SAGE layer: hout = relu( (agg/cnt) @ Wl + hin @ Wr + b )
// block = 128 threads = 4 waves; each wave owns 16 nodes.
// Staging: raw fp32 copies via async global->LDS (ASYNCcnt), normalize in regs.
__global__ __launch_bounds__(128) void k_sage(
    const float* __restrict__ hin, const float* __restrict__ agg,
    const float* __restrict__ cnt, const _Float16* __restrict__ wlf,
    const _Float16* __restrict__ wrf, const float* __restrict__ bias,
    float* __restrict__ hout) {
  __shared__ float lhf[64 * 64];  // self features, fp32 (16KB)
  __shared__ float lgf[64 * 64];  // raw aggregate,  fp32 (16KB)
  const int tid = threadIdx.x;
  const int nb  = blockIdx.x * 64;

  // Async-copy 64 nodes x 64 fp32 features of both operands into LDS.
  // Tail block reads a little past the logical end of hin/agg; both live in
  // the middle of the workspace allocation, so the reads are in-bounds memory
  // and the garbage rows never get stored (guarded epilogue).
  {
    uint32_t ldsH = (uint32_t)(uintptr_t)&lhf[0];
    uint32_t ldsG = (uint32_t)(uintptr_t)&lgf[0];
    const float* gh = hin + (size_t)nb * 64;
    const float* gg = agg + (size_t)nb * 64;
#pragma unroll
    for (int it = 0; it < 8; ++it) {
      uint32_t boff = (uint32_t)(tid + it * 128) * 16u;  // 1024 x b128 per matrix
      asm volatile("global_load_async_to_lds_b128 %0, %1, %2"
                   :: "v"(ldsH + boff), "v"(boff), "s"(gh) : "memory");
      asm volatile("global_load_async_to_lds_b128 %0, %1, %2"
                   :: "v"(ldsG + boff), "v"(boff), "s"(gg) : "memory");
    }
  }
  asm volatile("s_wait_asynccnt 0x0" ::: "memory");
  __syncthreads();

  const int lane = tid & 31, wave = tid >> 5;
  const int m  = lane & 15;
  const int ko = (lane & 16) ? 8 : 0;  // A-matrix: lanes 16-31 carry K+8 block
  const int myrow = wave * 16 + m;
  const float* rh = &lhf[myrow * 64];
  const float* rg = &lgf[myrow * 64];
  // mean-normalization factor for this lane's node row
  int gnode = nb + myrow;
  float rcp = 1.f / fmaxf(cnt[gnode < NN ? gnode : (NN - 1)], 1.f);

  v16h ah0, ah1, ag0, ag1;  // A fragments: {h,agg} x K-step {0,1}
#pragma unroll
  for (int j = 0; j < 8; ++j) {
    ah0[j]     = (_Float16)rh[ko + j];
    ah0[8 + j] = (_Float16)rh[16 + ko + j];
    ah1[j]     = (_Float16)rh[32 + ko + j];
    ah1[8 + j] = (_Float16)rh[48 + ko + j];
    ag0[j]     = (_Float16)(rg[ko + j] * rcp);
    ag0[8 + j] = (_Float16)(rg[16 + ko + j] * rcp);
    ag1[j]     = (_Float16)(rg[32 + ko + j] * rcp);
    ag1[8 + j] = (_Float16)(rg[48 + ko + j] * rcp);
  }

  const bool full = (nb + 64 <= NN);               // uniform: all but last block
  const int rbase = nb + wave * 16 + ((lane & 16) ? 8 : 0);  // D-row base
#pragma unroll
  for (int nt = 0; nt < 4; ++nt) {
    v8f acc = {};
    v16h b;
    b = *(const v16h*)&wlf[((0 * 4 + nt) * 32 + lane) * 16];
    acc = wmma16(ag0, b, acc);
    b = *(const v16h*)&wlf[((1 * 4 + nt) * 32 + lane) * 16];
    acc = wmma16(ag1, b, acc);
    b = *(const v16h*)&wrf[((0 * 4 + nt) * 32 + lane) * 16];
    acc = wmma16(ah0, b, acc);
    b = *(const v16h*)&wrf[((1 * 4 + nt) * 32 + lane) * 16];
    acc = wmma16(ah1, b, acc);

    int col = nt * 16 + m;
    float bv = bias[col];
    if (full) {
#pragma unroll
      for (int v = 0; v < 8; ++v)
        hout[(rbase + v) * 64 + col] = fmaxf(acc[v] + bv, 0.f);
    } else {
#pragma unroll
      for (int v = 0; v < 8; ++v) {
        int node = rbase + v;
        if (node < NN) hout[node * 64 + col] = fmaxf(acc[v] + bv, 0.f);
      }
    }
  }
}

// global mean pool scatter: one lane = (node, 4 features)
__global__ void k_pool(const float* __restrict__ h, const int* __restrict__ batch,
                       float* __restrict__ psum, float* __restrict__ pcnt) {
  int i = blockIdx.x * blockDim.x + threadIdx.x;
  if (i >= NN * 16) return;
  int node = i >> 4, q = i & 15;
  int g = batch[node];
  float4 v = ((const float4*)h)[i];
  float* p = &psum[g * 64 + q * 4];
  atomicAdd(p + 0, v.x);
  atomicAdd(p + 1, v.y);
  atomicAdd(p + 2, v.z);
  atomicAdd(p + 3, v.w);
  if (q == 0) atomicAdd(&pcnt[g], 1.f);
}

// classifier: out[g] = (psum[g]/pcnt[g]) @ wc + bc  via 2 chained WMMAs per 16 graphs
__global__ __launch_bounds__(32) void k_cls(
    const float* __restrict__ psum, const float* __restrict__ pcnt,
    const _Float16* __restrict__ wcf, const float* __restrict__ bc,
    float* __restrict__ out) {
  const int lane = threadIdx.x & 31;
  const int tile = blockIdx.x;           // 0..31 (512/16)
  const int m  = lane & 15;
  const int ko = (lane & 16) ? 8 : 0;
  const int g  = tile * 16 + m;
  const float rcp = 1.f / fmaxf(pcnt[g], 1.f);
  const float* row = &psum[g * 64];

  v16h a0, a1;
#pragma unroll
  for (int j = 0; j < 8; ++j) {
    a0[j]     = (_Float16)(row[ko + j] * rcp);
    a0[8 + j] = (_Float16)(row[16 + ko + j] * rcp);
    a1[j]     = (_Float16)(row[32 + ko + j] * rcp);
    a1[8 + j] = (_Float16)(row[48 + ko + j] * rcp);
  }
  v8f acc = {};
  v16h b0 = *(const v16h*)&wcf[(0 * 32 + lane) * 16];
  acc = wmma16(a0, b0, acc);
  v16h b1 = *(const v16h*)&wcf[(1 * 32 + lane) * 16];
  acc = wmma16(a1, b1, acc);

  int c = lane & 15;
  if (c < NCLS) {
    float bv = bc[c];
    int gb = tile * 16 + ((lane & 16) ? 8 : 0);
#pragma unroll
    for (int v = 0; v < 8; ++v) out[(gb + v) * NCLS + c] = acc[v] + bv;
  }
}

extern "C" void kernel_launch(void* const* d_in, const int* in_sizes, int n_in,
                              void* d_out, int out_size, void* d_ws, size_t ws_size,
                              hipStream_t stream) {
  (void)in_sizes; (void)n_in; (void)out_size; (void)ws_size;
  const int*   x     = (const int*)d_in[0];
  const int*   ei    = (const int*)d_in[1];
  const int*   batch = (const int*)d_in[2];
  // d_in[3] = num_graphs (compile-time NG here)
  const float* se  = (const float*)d_in[4];
  const float* ce  = (const float*)d_in[5];
  const float* w1l = (const float*)d_in[6];
  const float* w1r = (const float*)d_in[7];
  const float* b1  = (const float*)d_in[8];
  const float* w2l = (const float*)d_in[9];
  const float* w2r = (const float*)d_in[10];
  const float* b2  = (const float*)d_in[11];
  const float* wc  = (const float*)d_in[12];
  const float* bc  = (const float*)d_in[13];
  const int* src = ei;
  const int* tgt = ei + NE;

  char* ws = (char*)d_ws;
  size_t off = 0;
  auto alloc = [&](size_t bytes) -> void* {
    void* p = (void*)(ws + off);
    off += (bytes + 255) & ~(size_t)255;
    return p;
  };
  float* hA   = (float*)alloc(sizeof(float) * NN * 64);
  float* hC   = (float*)alloc(sizeof(float) * NN * 64);
  float* agg  = (float*)alloc(sizeof(float) * NN * 64);
  float* cnt  = (float*)alloc(sizeof(float) * NN);
  float* psum = (float*)alloc(sizeof(float) * NG * 64);
  float* pcnt = (float*)alloc(sizeof(float) * NG);
  _Float16* f1l = (_Float16*)alloc(sizeof(_Float16) * 4096);
  _Float16* f1r = (_Float16*)alloc(sizeof(_Float16) * 4096);
  _Float16* f2l = (_Float16*)alloc(sizeof(_Float16) * 4096);
  _Float16* f2r = (_Float16*)alloc(sizeof(_Float16) * 4096);
  _Float16* fwc = (_Float16*)alloc(sizeof(_Float16) * 1024);

  // weight packing (tiny)
  k_pack64<<<32, 128, 0, stream>>>(w1l, f1l);
  k_pack64<<<32, 128, 0, stream>>>(w1r, f1r);
  k_pack64<<<32, 128, 0, stream>>>(w2l, f2l);
  k_pack64<<<32, 128, 0, stream>>>(w2r, f2r);
  k_packC<<<8, 128, 0, stream>>>(wc, fwc);

  const int NF = NN * 64;
  k_embed<<<(NF + 255) / 256, 256, 0, stream>>>(x, se, ce, hA);

  // layer 1
  k_zero<<<(NF + 255) / 256, 256, 0, stream>>>(agg, NF);
  k_zero<<<(NN + 255) / 256, 256, 0, stream>>>(cnt, NN);
  k_scatter<<<(NE * 16) / 256, 256, 0, stream>>>(hA, src, tgt, agg, cnt);
  k_sage<<<(NN + 63) / 64, 128, 0, stream>>>(hA, agg, cnt, f1l, f1r, b1, hC);

  // layer 2
  k_zero<<<(NF + 255) / 256, 256, 0, stream>>>(agg, NF);
  k_zero<<<(NN + 255) / 256, 256, 0, stream>>>(cnt, NN);
  k_scatter<<<(NE * 16) / 256, 256, 0, stream>>>(hC, src, tgt, agg, cnt);
  k_sage<<<(NN + 63) / 64, 128, 0, stream>>>(hC, agg, cnt, f2l, f2r, b2, hA);

  // pool + classify
  k_zero<<<(NG * 64 + 255) / 256, 256, 0, stream>>>(psum, NG * 64);
  k_zero<<<(NG + 255) / 256, 256, 0, stream>>>(pcnt, NG);
  k_pool<<<(NN * 16 + 255) / 256, 256, 0, stream>>>(hA, batch, psum, pcnt);
  k_cls<<<32, 32, 0, stream>>>(psum, pcnt, fwc, bc, (float*)d_out);
}
